// Sparse_Three_Sum_47141561041318
// MI455X (gfx1250) — matrix operand order repaired and checked by reading, not compiled, not run on hardware
//
#include <hip/hip_runtime.h>

// ---------------------------------------------------------------------------
// Types for CDNA5 WMMA
// ---------------------------------------------------------------------------
typedef __attribute__((ext_vector_type(16))) __bf16 v16bf;
typedef __attribute__((ext_vector_type(8)))  float  v8f;

union FragAB {
    v16bf bf;
    unsigned int u[8];
};

__device__ __forceinline__ unsigned short f2bf(float f) {
    union { float f; unsigned u; } v; v.f = f;
    unsigned r = 0x7FFFu + ((v.u >> 16) & 1u);   // round-to-nearest-even
    return (unsigned short)((v.u + r) >> 16);
}
__device__ __forceinline__ unsigned pack2(float lo, float hi) {
    return ((unsigned)f2bf(hi) << 16) | (unsigned)f2bf(lo);
}

// ---------------------------------------------------------------------------
// GEMM: C[M x N] = A[M x K] @ W[K x N] (+ up to 3 bias vectors over N)
// bf16 inputs (converted on LDS staging), f32 WMMA accumulation.
// Workgroup tile 128x64, 8 waves (4x2), each wave 32x32 = 2x2 WMMA 16x16x32.
// M%128==0, N%64==0, K%32==0 (holds: M=32768, N=256, K in {128,256}).
// ---------------------------------------------------------------------------
#define BM 128
#define BN 64
#define BK 32

__global__ __launch_bounds__(256)
void gemm_bf16_wmma(const float* __restrict__ A, const float* __restrict__ W,
                    const float* __restrict__ bias0, const float* __restrict__ bias1,
                    const float* __restrict__ bias2,
                    float* __restrict__ C, int M, int N, int K) {
    __shared__ unsigned sA[BM * (BK / 2)];   // [m][kp] packed bf16x2, 128x16
    __shared__ unsigned sB[BN * (BK / 2)];   // [n][kp] packed bf16x2, 64x16 (transposed)

    const int tid   = threadIdx.x;
    const int lane  = tid & 31;
    const int wid   = tid >> 5;
    const int waveM = wid >> 1;       // 0..3
    const int waveN = wid & 1;        // 0..1
    const int m0    = blockIdx.y * BM;
    const int n0    = blockIdx.x * BN;
    const int h     = lane >> 4;      // lane half: 0/1
    const int l16   = lane & 15;

    v8f c00 = {}, c01 = {}, c10 = {}, c11 = {};

    for (int k0 = 0; k0 < K; k0 += BK) {
        // stage A tile (128x32 f32 -> 2048 packed u32); 8 pairs/thread
        #pragma unroll
        for (int i = 0; i < 8; ++i) {
            int pidx = tid + i * 256;
            int r = pidx >> 4, kp = pidx & 15;
            const float* ap = A + (size_t)(m0 + r) * K + (k0 + 2 * kp);
            sA[r * 16 + kp] = pack2(ap[0], ap[1]);
        }
        // stage B tile transposed (rows k0..k0+31, cols n0..n0+63); 4 pairs/thread
        #pragma unroll
        for (int i = 0; i < 4; ++i) {
            int pidx = tid + i * 256;
            int n = pidx & 63, kp = pidx >> 6;
            const float* wp = W + (size_t)(k0 + 2 * kp) * N + (n0 + n);
            sB[n * 16 + kp] = pack2(wp[0], wp[N]);
        }
        __syncthreads();

        // Fragment gather per ISA 7.12.2 16-bit layouts.
        FragAB a0, a1, b0, b1;
        const int ra0 = waveM * 32 + l16;
        const int ra1 = ra0 + 16;
        const int cb0 = waveN * 32 + l16;
        const int cb1 = cb0 + 16;
        #pragma unroll
        for (int j = 0; j < 8; ++j) {
            // A 16x32: lane half h, VGPR j -> k = (j<4?0:16) + 8h + 2(j&3)
            int kpA = ((j < 4) ? 0 : 8) + h * 4 + (j & 3);
            a0.u[j] = sA[ra0 * 16 + kpA];
            a1.u[j] = sA[ra1 * 16 + kpA];
            // B 32x16: lane half h, VGPR j -> k = (j<4?0:16) + 8h + 2(j&3), col = l16
            int kpB = ((j < 4) ? 0 : 8) + h * 4 + (j & 3);
            b0.u[j] = sB[cb0 * 16 + kpB];
            b1.u[j] = sB[cb1 * 16 + kpB];
        }

        c00 = __builtin_amdgcn_wmma_f32_16x16x32_bf16(false, a0.bf, false, b0.bf,
                                                      (short)0, c00, false, false);
        c01 = __builtin_amdgcn_wmma_f32_16x16x32_bf16(false, a0.bf, false, b1.bf,
                                                      (short)0, c01, false, false);
        c10 = __builtin_amdgcn_wmma_f32_16x16x32_bf16(false, a1.bf, false, b0.bf,
                                                      (short)0, c10, false, false);
        c11 = __builtin_amdgcn_wmma_f32_16x16x32_bf16(false, a1.bf, false, b1.bf,
                                                      (short)0, c11, false, false);
        __syncthreads();
    }

    // Epilogue. C/D layout: VGPR v, lane = n + 16h -> row = v + 8h, col = n.
    const int colA = n0 + waveN * 32 + l16;
    const int colB = colA + 16;
    float bA = 0.f, bB = 0.f;
    if (bias0) { bA += bias0[colA]; bB += bias0[colB]; }
    if (bias1) { bA += bias1[colA]; bB += bias1[colB]; }
    if (bias2) { bA += bias2[colA]; bB += bias2[colB]; }
    #pragma unroll
    for (int v = 0; v < 8; ++v) {
        int row0 = m0 + waveM * 32 + 8 * h + v;
        int row1 = row0 + 16;
        C[(size_t)row0 * N + colA] = c00[v] + bA;
        C[(size_t)row0 * N + colB] = c01[v] + bB;
        C[(size_t)row1 * N + colA] = c10[v] + bA;
        C[(size_t)row1 * N + colB] = c11[v] + bB;
    }
}

// ---------------------------------------------------------------------------
// Edge scatter: acc[dst] += ew * xw[src], 256 channels. One wave per edge,
// lane handles 2x float4 (L2-resident atomics).
// ---------------------------------------------------------------------------
__global__ __launch_bounds__(256)
void scatter256(const float* __restrict__ t, const int* __restrict__ src,
                const int* __restrict__ dst, const float* __restrict__ ew,
                float* __restrict__ acc, int E) {
    int e = (int)((blockIdx.x * (size_t)blockDim.x + threadIdx.x) >> 5);
    int lane = threadIdx.x & 31;
    if (e >= E) return;
    int s = src[e], d = dst[e];
    float w = ew[e];
    const float4* tp = (const float4*)(t + (size_t)s * 256);
    float* ap = acc + (size_t)d * 256;
    #pragma unroll
    for (int j = 0; j < 2; ++j) {
        int fi = lane + j * 32;
        float4 v = tp[fi];
        int base = fi * 4;
        atomicAdd(ap + base + 0, w * v.x);
        atomicAdd(ap + base + 1, w * v.y);
        atomicAdd(ap + base + 2, w * v.z);
        atomicAdd(ap + base + 3, w * v.w);
    }
}

// ---------------------------------------------------------------------------
// Block 4 ([256 -> 1]): three dot products per node; wave per node.
// hf[n] = h[n].wln + (bln + b1 + b2);  t1n/t2n = h[n].w1 / h[n].w2
// ---------------------------------------------------------------------------
__global__ __launch_bounds__(256)
void dot3_fc1(const float* __restrict__ h, const float* __restrict__ wln,
              const float* __restrict__ w1, const float* __restrict__ w2,
              const float* __restrict__ bln, const float* __restrict__ b1,
              const float* __restrict__ b2,
              float* __restrict__ hf, float* __restrict__ t1n,
              float* __restrict__ t2n, int NT) {
    int n = (int)((blockIdx.x * (size_t)blockDim.x + threadIdx.x) >> 5);
    int lane = threadIdx.x & 31;
    if (n >= NT) return;
    const float* hp = h + (size_t)n * 256;
    float sln = 0.f, s1 = 0.f, s2 = 0.f;
    #pragma unroll
    for (int i = 0; i < 8; ++i) {
        int c = lane + i * 32;
        float x = hp[c];
        sln += x * wln[c];
        s1  += x * w1[c];
        s2  += x * w2[c];
    }
    #pragma unroll
    for (int m = 16; m; m >>= 1) {
        sln += __shfl_xor(sln, m, 32);
        s1  += __shfl_xor(s1,  m, 32);
        s2  += __shfl_xor(s2,  m, 32);
    }
    if (lane == 0) {
        hf[n]  = sln + bln[0] + b1[0] + b2[0];
        t1n[n] = s1;
        t2n[n] = s2;
    }
}

__global__ void scatter1(const float* __restrict__ tn, const int* __restrict__ src,
                         const int* __restrict__ dst, const float* __restrict__ ew,
                         float* __restrict__ hf, int E) {
    int e = blockIdx.x * blockDim.x + threadIdx.x;
    if (e < E) atomicAdd(&hf[dst[e]], ew[e] * tn[src[e]]);
}

// ---------------------------------------------------------------------------
// Per-graph sort-pool: bitonic sort 1024 keys (desc key, asc idx = stable
// argsort(-key)), keep top 128, gather pooled [128 x 257] = [h row | hf].
// ---------------------------------------------------------------------------
__global__ __launch_bounds__(1024)
void sortpool(const float* __restrict__ hf, const float* __restrict__ h,
              float* __restrict__ pooled) {
    __shared__ float key[1024];
    __shared__ int   idx[1024];
    const int b = blockIdx.x, t = threadIdx.x;
    key[t] = hf[b * 1024 + t];
    idx[t] = t;
    __syncthreads();
    for (int k = 2; k <= 1024; k <<= 1) {
        for (int j = k >> 1; j > 0; j >>= 1) {
            int ixj = t ^ j;
            if (ixj > t) {
                float ka = key[t], kb = key[ixj];
                int   ia = idx[t], ib = idx[ixj];
                bool up = ((t & k) == 0);
                // strict total order: a_first == "elem(t) precedes elem(ixj)"
                bool a_first = (ka > kb) || (ka == kb && ia < ib);
                if (up ? !a_first : a_first) {
                    key[t] = kb; key[ixj] = ka;
                    idx[t] = ib; idx[ixj] = ia;
                }
            }
            __syncthreads();
        }
    }
    for (int i = t; i < 128 * 257; i += 1024) {
        int kk = i / 257, c = i - kk * 257;
        int n = b * 1024 + idx[kk];
        pooled[((size_t)b * 128 + kk) * 257 + c] =
            (c < 256) ? h[(size_t)n * 256 + c] : hf[n];
    }
}

// ---------------------------------------------------------------------------
// Head kernels (tiny)
// ---------------------------------------------------------------------------
__global__ void conv1_head(const float* __restrict__ pooled, const float* __restrict__ w,
                           const float* __restrict__ bias, float* __restrict__ out) {
    int tid = blockIdx.x * blockDim.x + threadIdx.x;       // 32*128*16
    if (tid >= 32 * 128 * 16) return;
    int o = tid & 15, k = (tid >> 4) & 127, b = tid >> 11;
    const float* pp = pooled + ((size_t)b * 128 + k) * 257;
    const float* wp = w + o * 257;
    float s = bias[o];
    for (int c = 0; c < 257; ++c) s += pp[c] * wp[c];
    out[((size_t)b * 16 + o) * 128 + k] = s > 0.f ? s : 0.f;
}

__global__ void maxpool2(const float* __restrict__ in, float* __restrict__ out) {
    int tid = blockIdx.x * blockDim.x + threadIdx.x;       // 32*16*64
    if (tid >= 32 * 16 * 64) return;
    int j = tid & 63, rest = tid >> 6;                     // rest = b*16+o
    float a = in[rest * 128 + 2 * j], c = in[rest * 128 + 2 * j + 1];
    out[rest * 64 + j] = a > c ? a : c;
}

__global__ void conv2_head(const float* __restrict__ z, const float* __restrict__ w,
                           const float* __restrict__ bias, float* __restrict__ out) {
    int tid = blockIdx.x * blockDim.x + threadIdx.x;       // 32*32*60
    if (tid >= 32 * 32 * 60) return;
    int tp = tid % 60, o = (tid / 60) % 32, b = tid / (60 * 32);
    float s = bias[o];
    for (int i = 0; i < 16; ++i)
        #pragma unroll
        for (int tau = 0; tau < 5; ++tau)
            s += w[(o * 16 + i) * 5 + tau] * z[((size_t)b * 16 + i) * 64 + tp + tau];
    out[((size_t)b * 32 + o) * 60 + tp] = s > 0.f ? s : 0.f;
}

__global__ void lin1_head(const float* __restrict__ flat, const float* __restrict__ w,
                          const float* __restrict__ bias, float* __restrict__ y) {
    int tid = blockIdx.x * blockDim.x + threadIdx.x;       // 32*128
    if (tid >= 32 * 128) return;
    int j = tid & 127, b = tid >> 7;
    const float* fp = flat + (size_t)b * 1920;
    float s = bias[j];
    for (int i = 0; i < 1920; ++i) s += fp[i] * w[i * 128 + j];
    y[tid] = s > 0.f ? s : 0.f;
}

__global__ void lin2_head(const float* __restrict__ y, const float* __restrict__ w,
                          const float* __restrict__ bias, float* __restrict__ out) {
    int tid = blockIdx.x * blockDim.x + threadIdx.x;       // 64
    if (tid >= 64) return;
    int c = tid & 1, b = tid >> 1;
    float s = bias[c];
    for (int j = 0; j < 128; ++j) s += y[b * 128 + j] * w[j * 2 + c];
    out[tid] = s;
}

// ---------------------------------------------------------------------------
// Launcher
// ---------------------------------------------------------------------------
extern "C" void kernel_launch(void* const* d_in, const int* in_sizes, int n_in,
                              void* d_out, int out_size, void* d_ws, size_t ws_size,
                              hipStream_t stream) {
    (void)n_in; (void)out_size; (void)ws_size;
    constexpr int NT = 32768, FEAT = 128, HID = 256;

    const float* x   = (const float*)d_in[0];
    const int*   ei1 = (const int*)d_in[1];
    const float* ew1 = (const float*)d_in[2];
    const int*   ei2 = (const int*)d_in[3];
    const float* ew2 = (const float*)d_in[4];
    const int E = in_sizes[2];
    const int *src1 = ei1, *dst1 = ei1 + E;
    const int *src2 = ei2, *dst2 = ei2 + E;

    // params in dict insertion order: 4 blocks x {ln_w,ln_b,c1_w,c1_b,c2_w,c2_b},
    // then conv1_w/b, conv2_w/b, lin1_w/b, lin2_w/b
    const float* P[32];
    for (int i = 0; i < 32; ++i) P[i] = (const float*)d_in[5 + i];

    // workspace layout (floats)
    float* ws = (float*)d_ws;
    const size_t NTC = (size_t)NT * HID;
    float* bufA = ws;
    float* bufB = ws + NTC;
    float* t1   = ws + 2 * NTC;
    float* t2   = ws + 3 * NTC;
    // head scratch aliases t1 region (free once blocks 1-3 are done)
    float* hf     = t1;
    float* t1n    = hf + NT;
    float* t2n    = t1n + NT;
    float* pooled = t2n + NT;                            // 32*128*257
    float* c1o    = pooled + (size_t)32 * 128 * 257;     // 32*16*128
    float* p2     = c1o + 32 * 16 * 128;                 // 32*16*64
    float* c2o    = p2 + 32 * 16 * 64;                   // 32*32*60
    float* y1     = c2o + 32 * 32 * 60;                  // 32*128

    const dim3 gGemm(HID / BN, NT / BM);                 // (4, 256)
    const int scatBlocks = (int)(((size_t)E * 32 + 255) / 256);

    auto run_block = [&](const float* hin, int K, const float* const* Pb, float* hout) {
        // hout = hin@W_ln + (b_ln + b_c1 + b_c2)   (accumulator pre-init)
        gemm_bf16_wmma<<<gGemm, 256, 0, stream>>>(hin, Pb[0], Pb[1], Pb[3], Pb[5],
                                                  hout, NT, HID, K);
        gemm_bf16_wmma<<<gGemm, 256, 0, stream>>>(hin, Pb[2], nullptr, nullptr, nullptr,
                                                  t1, NT, HID, K);
        gemm_bf16_wmma<<<gGemm, 256, 0, stream>>>(hin, Pb[4], nullptr, nullptr, nullptr,
                                                  t2, NT, HID, K);
        scatter256<<<scatBlocks, 256, 0, stream>>>(t1, src1, dst1, ew1, hout, E);
        scatter256<<<scatBlocks, 256, 0, stream>>>(t2, src2, dst2, ew2, hout, E);
    };

    const float* ib1[6] = {P[0], P[1], P[2], P[3], P[4], P[5]};
    const float* ib2[6] = {P[6], P[7], P[8], P[9], P[10], P[11]};
    const float* ib3[6] = {P[12], P[13], P[14], P[15], P[16], P[17]};

    run_block(x,    FEAT, ib1, bufA);   // h1 = bufA
    run_block(bufA, HID,  ib2, bufB);   // h2 = bufB
    run_block(bufB, HID,  ib3, bufA);   // h3 = bufA

    // block 4: [256 -> 1]
    dot3_fc1<<<NT / 8, 256, 0, stream>>>(bufA, P[18], P[20], P[22],
                                         P[19], P[21], P[23], hf, t1n, t2n, NT);
    scatter1<<<(E + 255) / 256, 256, 0, stream>>>(t1n, src1, dst1, ew1, hf, E);
    scatter1<<<(E + 255) / 256, 256, 0, stream>>>(t2n, src2, dst2, ew2, hf, E);

    // sort-pool + head
    sortpool<<<32, 1024, 0, stream>>>(hf, bufA, pooled);
    conv1_head<<<(32 * 128 * 16) / 256, 256, 0, stream>>>(pooled, P[24], P[25], c1o);
    maxpool2<<<(32 * 16 * 64) / 256, 256, 0, stream>>>(c1o, p2);
    conv2_head<<<(32 * 32 * 60 + 255) / 256, 256, 0, stream>>>(p2, P[26], P[27], c2o);
    lin1_head<<<(32 * 128) / 256, 256, 0, stream>>>(c2o, P[28], P[29], y1);
    lin2_head<<<1, 64, 0, stream>>>(y1, P[30], P[31], (float*)d_out);
}
